// ManualTuckerLinear_6150393168457
// MI455X (gfx1250) — compile-verified, matrix operand order checked by reading, not run on hardware
//
#include <hip/hip_runtime.h>
#include <hip/hip_bf16.h>

// ---------------------------------------------------------------------------
// Problem constants (from reference): x[B,S,N] fp16, out[B,S,M] fp16
// ---------------------------------------------------------------------------
#define TOK   16384      // B*S = 4*4096
#define NDIM  1152
#define MDIM  4608
#define RDIM  64
#define LRDIM 32
#define KHL   96         // R + LR, fused second-stage K
#define WCAT_BYTES (MDIM * KHL * 2)   // 884736, 256B-aligned already

typedef __attribute__((ext_vector_type(16))) _Float16 v16h;
typedef __attribute__((ext_vector_type(8)))  _Float16 v8h;
typedef __attribute__((ext_vector_type(8)))  float    v8f;

static __device__ __forceinline__ v8f wmma_f16(v16h a, v16h b, v8f c) {
    // D = A(16x32 f16) * B(32x16 f16) + C(16x16 f32)
    return __builtin_amdgcn_wmma_f32_16x16x32_f16(
        /*neg_a=*/false, a, /*neg_b=*/false, b,
        /*c_mod=*/(short)0, c, /*reuse_a=*/false, /*reuse_b=*/false);
}

// Load a 16x32 f16 fragment (A layout; identical code serves B since B's
// lane->column mapping mirrors A's lane->row mapping).
// ISA layout: lanes 0-15 row m=lane, k in {0..7, 16..23};
//             lanes 16-31 row m=lane-16, k in {8..15, 24..31}.
// base points at element [0][0] of a row-major [16][stride] tile (halves).
static __device__ __forceinline__ v16h load_frag(const _Float16* __restrict__ base,
                                                 int stride, int lane) {
    int m  = lane & 15;
    int k0 = (lane >> 4) << 3;           // 0 or 8
    const _Float16* p = base + (size_t)m * stride + k0;
    v8h lo = *(const v8h*)(p);           // k0 .. k0+7     (16B aligned)
    v8h hi = *(const v8h*)(p + 16);      // k0+16 .. k0+23 (16B aligned)
    return __builtin_shufflevector(lo, hi, 0,1,2,3,4,5,6,7,8,9,10,11,12,13,14,15);
}

// NVFP4 (E2M1) nearest-magnitude snap, searchsorted-left boundary semantics.
static __device__ __forceinline__ float e2m1_snap(float t) {
    float a = __builtin_fabsf(t);
    float q;
    if      (a <= 0.25f) q = 0.0f;
    else if (a <= 0.75f) q = 0.5f;
    else if (a <= 1.25f) q = 1.0f;
    else if (a <= 1.75f) q = 1.5f;
    else if (a <= 2.5f)  q = 2.0f;
    else if (a <= 3.5f)  q = 3.0f;
    else if (a <= 5.0f)  q = 4.0f;
    else                 q = 6.0f;
    return __builtin_copysignf(q, t);
}

// ---------------------------------------------------------------------------
// Kernel A: Wcat[m][0:64] = (U1 @ G)[m][:],  Wcat[m][64:96] = lora_b[m][:]
// ---------------------------------------------------------------------------
__global__ void k_build_wcat(const _Float16* __restrict__ u1,
                             const _Float16* __restrict__ g,
                             const _Float16* __restrict__ lb,
                             _Float16* __restrict__ wcat) {
    int idx = blockIdx.x * 256 + threadIdx.x;
    if (idx >= MDIM * KHL) return;
    int m = idx / KHL, c = idx % KHL;
    if (c < RDIM) {
        float acc = 0.f;
#pragma unroll 8
        for (int q = 0; q < RDIM; ++q)
            acc += (float)u1[m * RDIM + q] * (float)g[q * RDIM + c];
        wcat[idx] = (_Float16)acc;
    } else {
        wcat[idx] = lb[m * LRDIM + (c - RDIM)];
    }
}

// ---------------------------------------------------------------------------
// Kernel B: per 128-token tile: smooth + NVFP4 quant-dequant, then
//   hl[t][0:64]  = x_q     @ u2_q^T   (K=N loop, WMMA f16)
//   hl[t][64:96] = x_smooth@ lora_a^T
// 256 threads = 8 waves, wave w owns tokens [t0+16w, t0+16w+16).
// ---------------------------------------------------------------------------
__global__ void k_stage1(const _Float16* __restrict__ x,
                         const _Float16* __restrict__ ss,
                         const _Float16* __restrict__ u2,
                         const _Float16* __restrict__ la,
                         _Float16* __restrict__ hl) {
    __shared__ _Float16 lds_q[8][16][32];
    __shared__ _Float16 lds_s[8][16][32];

    const int lane = threadIdx.x & 31;
    const int wave = threadIdx.x >> 5;
    const size_t t0 = (size_t)blockIdx.x * 128 + (size_t)wave * 16;

    v8f acc_h[4] = {};   // 16x64 h accumulator
    v8f acc_l[2] = {};   // 16x32 lora accumulator

    const int row = lane & 15;       // token row within wave tile
    const int blk = lane >> 4;       // which 16-wide quant block of the chunk

    for (int ch = 0; ch < NDIM / 32; ++ch) {
        const int nc = ch * 32;

        // ---- smooth + NVFP4 quant-dequant: one lane = one 16-block ----
        {
            const _Float16* xp = x + (t0 + row) * NDIM + nc + blk * 16;
            v8h xa = *(const v8h*)xp;
            v8h xb = *(const v8h*)(xp + 8);
            const _Float16* sp = ss + nc + blk * 16;
            v8h sa = *(const v8h*)sp;
            v8h sb = *(const v8h*)(sp + 8);

            float vs[16];
#pragma unroll
            for (int i = 0; i < 8; ++i) {
                vs[i]     = (float)xa[i] * (float)sa[i];
                vs[8 + i] = (float)xb[i] * (float)sb[i];
            }
            float amax = 0.f;
#pragma unroll
            for (int i = 0; i < 16; ++i)
                amax = __builtin_fmaxf(amax, __builtin_fabsf(vs[i]));
            float scale = __builtin_fmaxf(amax * (1.0f / 6.0f), 1e-6f);
            float inv   = 1.0f / scale;

            v8h qa, qb, ma, mb;
#pragma unroll
            for (int i = 0; i < 8; ++i) {
                ma[i] = (_Float16)vs[i];
                mb[i] = (_Float16)vs[8 + i];
                qa[i] = (_Float16)(e2m1_snap(vs[i]     * inv) * scale);
                qb[i] = (_Float16)(e2m1_snap(vs[8 + i] * inv) * scale);
            }
            _Float16* lq = &lds_q[wave][row][blk * 16];
            _Float16* ls = &lds_s[wave][row][blk * 16];
            *(v8h*)lq = qa;  *(v8h*)(lq + 8) = qb;
            *(v8h*)ls = ma;  *(v8h*)(ls + 8) = mb;
        }
        __syncthreads();

        // ---- WMMA accumulate ----
        v16h aq = load_frag(&lds_q[wave][0][0], 32, lane);
        v16h as = load_frag(&lds_s[wave][0][0], 32, lane);
#pragma unroll
        for (int j = 0; j < 4; ++j) {
            v16h bq = load_frag(u2 + (size_t)(j * 16) * NDIM + nc, NDIM, lane);
            acc_h[j] = wmma_f16(aq, bq, acc_h[j]);
        }
#pragma unroll
        for (int j = 0; j < 2; ++j) {
            v16h bl = load_frag(la + (size_t)(j * 16) * NDIM + nc, NDIM, lane);
            acc_l[j] = wmma_f16(as, bl, acc_l[j]);
        }
        __syncthreads();
    }

    // ---- write hl[T,96] (C layout: row = v + 8*(lane>=16), col = lane&15) ----
    const int mb = (lane >> 4) * 8;
    const int n  = lane & 15;
#pragma unroll
    for (int j = 0; j < 4; ++j)
#pragma unroll
        for (int v = 0; v < 8; ++v)
            hl[(t0 + mb + v) * KHL + j * 16 + n] = (_Float16)acc_h[j][v];
#pragma unroll
    for (int j = 0; j < 2; ++j)
#pragma unroll
        for (int v = 0; v < 8; ++v)
            hl[(t0 + mb + v) * KHL + 64 + j * 16 + n] = (_Float16)acc_l[j][v];
}

// ---------------------------------------------------------------------------
// Kernel C: out[t][m] = sum_k hl[t][k] * Wcat[m][k] + bias[m]   (K = 96)
// grid.x: 128-token groups, grid.y: 16 M-chunks (9 pairs of 32 cols each).
// Output staged through LDS for fully coalesced 16B/lane stores.
// ---------------------------------------------------------------------------
__global__ void k_stage2(const _Float16* __restrict__ hl,
                         const _Float16* __restrict__ wcat,
                         const _Float16* __restrict__ bias,
                         _Float16* __restrict__ out) {
    __shared__ _Float16 lds[8][16][32];

    const int lane = threadIdx.x & 31;
    const int wave = threadIdx.x >> 5;
    const size_t t0 = (size_t)blockIdx.x * 128 + (size_t)wave * 16;

    // A fragments for this wave's 16 tokens (K = 96 -> 3 fragments), loaded once.
    v16h A0 = load_frag(hl + t0 * KHL + 0,  KHL, lane);
    v16h A1 = load_frag(hl + t0 * KHL + 32, KHL, lane);
    v16h A2 = load_frag(hl + t0 * KHL + 64, KHL, lane);

    const int mb = (lane >> 4) * 8;
    const int n  = lane & 15;

    for (int p = 0; p < 9; ++p) {
        const int m0 = (blockIdx.y * 9 + p) * 32;

        v8f a0 = {}, a1 = {};
        a0 = wmma_f16(A0, load_frag(wcat + (size_t)m0 * KHL + 0,  KHL, lane), a0);
        a0 = wmma_f16(A1, load_frag(wcat + (size_t)m0 * KHL + 32, KHL, lane), a0);
        a0 = wmma_f16(A2, load_frag(wcat + (size_t)m0 * KHL + 64, KHL, lane), a0);
        a1 = wmma_f16(A0, load_frag(wcat + (size_t)(m0 + 16) * KHL + 0,  KHL, lane), a1);
        a1 = wmma_f16(A1, load_frag(wcat + (size_t)(m0 + 16) * KHL + 32, KHL, lane), a1);
        a1 = wmma_f16(A2, load_frag(wcat + (size_t)(m0 + 16) * KHL + 64, KHL, lane), a1);

        float b0 = (float)bias[m0 + n];
        float b1 = (float)bias[m0 + 16 + n];

#pragma unroll
        for (int v = 0; v < 8; ++v) {
            lds[wave][mb + v][n]      = (_Float16)(a0[v] + b0);
            lds[wave][mb + v][16 + n] = (_Float16)(a1[v] + b1);
        }
        __syncthreads();

        // coalesced store: 16 rows x 32 cols f16 = 64 slots of 8 halves (16B)
#pragma unroll
        for (int s = 0; s < 2; ++s) {
            int slot = s * 32 + lane;
            int r  = slot >> 2;
            int cg = slot & 3;
            v8h d = *(const v8h*)&lds[wave][r][cg * 8];
            *(v8h*)(out + (t0 + r) * (size_t)MDIM + m0 + cg * 8) = d;
        }
        __syncthreads();
    }
}

// ---------------------------------------------------------------------------
extern "C" void kernel_launch(void* const* d_in, const int* in_sizes, int n_in,
                              void* d_out, int out_size, void* d_ws, size_t ws_size,
                              hipStream_t stream) {
    const _Float16* x    = (const _Float16*)d_in[0];   // [4,4096,1152]
    const _Float16* ss   = (const _Float16*)d_in[1];   // [1152]
    const _Float16* u2   = (const _Float16*)d_in[2];   // [64,1152]
    const _Float16* g    = (const _Float16*)d_in[3];   // [64,64]
    const _Float16* u1   = (const _Float16*)d_in[4];   // [4608,64]
    const _Float16* la   = (const _Float16*)d_in[5];   // [32,1152]
    const _Float16* lb   = (const _Float16*)d_in[6];   // [4608,32]
    const _Float16* bias = (const _Float16*)d_in[7];   // [4608]
    _Float16* out = (_Float16*)d_out;                  // [4,4096,4608]

    _Float16* wcat = (_Float16*)d_ws;                          // [4608,96]
    _Float16* hl   = (_Float16*)((char*)d_ws + WCAT_BYTES);    // [16384,96]

    // A: fold U1@G and concat lora_b -> Wcat
    k_build_wcat<<<(MDIM * KHL + 255) / 256, 256, 0, stream>>>(u1, g, lb, wcat);
    // B: smooth + NVFP4 quant + skinny GEMMs -> hl[T,96]
    k_stage1<<<TOK / 128, 256, 0, stream>>>(x, ss, u2, la, hl);
    // C: hl @ Wcat^T + bias -> out
    k_stage2<<<dim3(TOK / 128, 16), 256, 0, stream>>>(hl, wcat, bias, out);
}
